// NodeOnlyModel_75617194213648
// MI455X (gfx1250) — compile-verified, matrix operand order checked by reading, not run on hardware
//
#include <hip/hip_runtime.h>
#include <hip/hip_bf16.h>
#include <math.h>

// ---------------------------------------------------------------------------
// Graph-attention + edge-MLP for MI455X (gfx1250), wave32, bf16 WMMA.
// Workspace: (4*N*64 + E*64 + E*4 + N*72 + 16384) * 4 bytes  (~283 MB).
// ---------------------------------------------------------------------------

typedef __attribute__((ext_vector_type(16))) __bf16 v16bf;
typedef __attribute__((ext_vector_type(8)))  float  v8f;

#define WMMA_BF16(a, b, c) \
    __builtin_amdgcn_wmma_f32_16x16x32_bf16(false, (a), false, (b), (short)0, (c), false, false)

// ---- WMMA fragment helpers (16x16x32 bf16, wave32 layouts per ISA 7.12.2) --
// A 16x32 bf16: lane holds row = lane&15; elements 0..7 -> K = k0+8*(lane>>4)+i,
// elements 8..15 -> K = k0+16+8*(lane>>4)+(i-8).  Vectorized: two 32B-aligned
// float4 pairs per fragment -> 4x global_load_b128 + v_cvt_pk_bf16_f32.
__device__ __forceinline__ v16bf load_row_bf16(const float* __restrict__ rowPtr,
                                               int k0, int lane) {
    int base = k0 + ((lane >> 4) << 3);
    const float4* p0 = (const float4*)(rowPtr + base);
    const float4* p1 = (const float4*)(rowPtr + base + 16);
    float4 a0 = p0[0], a1 = p0[1];
    float4 a2 = p1[0], a3 = p1[1];
    v16bf f;
    f[0]  = (__bf16)a0.x; f[1]  = (__bf16)a0.y; f[2]  = (__bf16)a0.z; f[3]  = (__bf16)a0.w;
    f[4]  = (__bf16)a1.x; f[5]  = (__bf16)a1.y; f[6]  = (__bf16)a1.z; f[7]  = (__bf16)a1.w;
    f[8]  = (__bf16)a2.x; f[9]  = (__bf16)a2.y; f[10] = (__bf16)a2.z; f[11] = (__bf16)a2.w;
    f[12] = (__bf16)a3.x; f[13] = (__bf16)a3.y; f[14] = (__bf16)a3.z; f[15] = (__bf16)a3.w;
    return f;
}

// B 32x16 bf16 gathered from a row-major [K, ld] f32 weight (pack kernel only).
__device__ __forceinline__ v16bf load_b_colmajor(const float* __restrict__ B, int ld,
                                                 int col, int k0, int lane) {
    int base = k0 + ((lane >> 4) << 3);
    v16bf f;
#pragma unroll
    for (int i = 0; i < 8; ++i)  f[i] = (__bf16)B[(size_t)(base + i) * ld + col];
#pragma unroll
    for (int i = 8; i < 16; ++i) f[i] = (__bf16)B[(size_t)(base + 8 + i) * ld + col];
    return f;
}

// C/D 16x16 f32: col = lane&15, rows = row0 + 8*(lane>>4) + j.
__device__ __forceinline__ void store_tile(float* __restrict__ C, int ld, int row0,
                                           int col0, int lane, v8f c) {
    int col   = col0 + (lane & 15);
    int rbase = row0 + ((lane >> 4) << 3);
#pragma unroll
    for (int j = 0; j < 8; ++j) C[(size_t)(rbase + j) * ld + col] = c[j];
}

__device__ __forceinline__ unsigned f2ord(float f) {
    unsigned u = __float_as_uint(f);
    return (u & 0x80000000u) ? ~u : (u | 0x80000000u);
}
__device__ __forceinline__ float ord2f(unsigned u) {
    return __uint_as_float((u & 0x80000000u) ? (u & 0x7FFFFFFFu) : ~u);
}

static __device__ const v8f V8F_ZERO = {0.f, 0.f, 0.f, 0.f, 0.f, 0.f, 0.f, 0.f};

// Fragment load from a *kernarg* base pointer: after unrolling, `frag` is a
// compile-time constant -> global_load_b128 saddr + immediate offset.
__device__ __forceinline__ v16bf load_frag(const __bf16* __restrict__ wfrag,
                                           int frag, int lane) {
    return ((const v16bf*)wfrag)[frag * 32 + lane];
}

// ---------------------------------------------------------------------------
__global__ void zero_kernel(unsigned* __restrict__ p, long n) {
    long i = (long)blockIdx.x * blockDim.x + threadIdx.x;
    if (i < n) p[i] = 0u;
}

// ---------------------------------------------------------------------------
// Pack all weights into fragment-ordered bf16: fragment f, lane l -> 16
// contiguous bf16 at wpack[(f*32+l)*16]. Consumer load = one v16bf per lane.
// Fragment map: Wq: 0+t*2+s  Wk: 8+..  Wv: 16+..  We: 24+..  Wo: 32+..
//               W1: 40+t*6+s   (t = 16-col tile, s = 32-wide K step)
__global__ void pack_weights_kernel(const float* __restrict__ Wq, const float* __restrict__ Wk,
                                    const float* __restrict__ Wv, const float* __restrict__ We,
                                    const float* __restrict__ Wo, const float* __restrict__ W1,
                                    __bf16* __restrict__ wpack) {
    int lane = threadIdx.x & 31;
    int frag = blockIdx.x * (blockDim.x >> 5) + (threadIdx.x >> 5);
    if (frag >= 64) return;
    const float* src;
    int t, s;
    if (frag < 8)       { src = Wq; t = (frag)      >> 1; s = frag & 1; }
    else if (frag < 16) { src = Wk; t = (frag - 8)  >> 1; s = frag & 1; }
    else if (frag < 24) { src = Wv; t = (frag - 16) >> 1; s = frag & 1; }
    else if (frag < 32) { src = We; t = (frag - 24) >> 1; s = frag & 1; }
    else if (frag < 40) { src = Wo; t = (frag - 32) >> 1; s = frag & 1; }
    else                { int f = frag - 40; src = W1; t = f / 6; s = f % 6; }
    v16bf b = load_b_colmajor(src, 64, (t << 4) + (lane & 15), s * 32, lane);
    ((v16bf*)wpack)[frag * 32 + lane] = b;
}

// ---------------------------------------------------------------------------
// Generic C[M,64] = A[M,64] @ W.  `wfrag` points at the packed 64x64 weight
// (8 fragments); one 16-row M-tile per wave; 4 accumulators -> no spill.
__global__ void gemm64_kernel(const float* __restrict__ A, const __bf16* __restrict__ wfrag,
                              float* __restrict__ C, int ntiles) {
    int lane = threadIdx.x & 31;
    int tile = blockIdx.x * (blockDim.x >> 5) + (threadIdx.x >> 5);
    if (tile >= ntiles) return;
    int m0 = tile << 4;
    const float* rowPtr = A + (size_t)(m0 + (lane & 15)) * 64;

    v16bf a0 = load_row_bf16(rowPtr, 0, lane);
    v16bf a1 = load_row_bf16(rowPtr, 32, lane);

    v8f acc[4];
#pragma unroll
    for (int i = 0; i < 4; ++i) acc[i] = V8F_ZERO;
#pragma unroll
    for (int t = 0; t < 4; ++t) {
        acc[t] = WMMA_BF16(a0, load_frag(wfrag, t * 2 + 0, lane), acc[t]);
        acc[t] = WMMA_BF16(a1, load_frag(wfrag, t * 2 + 1, lane), acc[t]);
    }
#pragma unroll
    for (int t = 0; t < 4; ++t) store_tile(C, 64, m0, t << 4, lane, acc[t]);
}

// per-(edge,head) attention scores + segment max via ordered-uint atomicMax
__global__ void scores_kernel(const float* __restrict__ q, const float* __restrict__ k,
                              const float* __restrict__ ep,
                              const int* __restrict__ srcIdx, const int* __restrict__ tgtIdx,
                              float* __restrict__ sc, unsigned* __restrict__ smaxU, int total) {
    int gid = blockIdx.x * blockDim.x + threadIdx.x;
    if (gid >= total) return;
    int e = gid >> 2, h = gid & 3;
    int s = srcIdx[e], t = tgtIdx[e];
    const float4* qp  = (const float4*)(q  + (size_t)t * 64 + (h << 4));
    const float4* kp  = (const float4*)(k  + (size_t)s * 64 + (h << 4));
    const float4* epp = (const float4*)(ep + (size_t)e * 64 + (h << 4));
    float a = 0.f;
#pragma unroll
    for (int i = 0; i < 4; ++i) {
        float4 qq = qp[i], kk = kp[i], ee = epp[i];
        a += qq.x * (kk.x + ee.x) + qq.y * (kk.y + ee.y) +
             qq.z * (kk.z + ee.z) + qq.w * (kk.w + ee.w);
    }
    a *= 0.25f;  // 1/sqrt(DH), DH=16
    sc[gid] = a;
    atomicMax(&smaxU[(size_t)t * 4 + h], f2ord(a));
}

// ex = exp(score - smax[tgt]); denom += ex  (in-place over sc)
__global__ void exp_kernel(const int* __restrict__ tgtIdx,
                           const unsigned* __restrict__ smaxU,
                           float* __restrict__ sc, float* __restrict__ denom, int total) {
    int gid = blockIdx.x * blockDim.x + threadIdx.x;
    if (gid >= total) return;
    int e = gid >> 2, h = gid & 3;
    int t = tgtIdx[e];
    float m  = ord2f(smaxU[(size_t)t * 4 + h]);
    float ex = __expf(sc[gid] - m);
    sc[gid] = ex;
    atomicAdd(&denom[(size_t)t * 4 + h], ex);
}

// agg[tgt] += alpha * (v[src] + ep), one thread per (edge, 4-dim group)
__global__ void msg_kernel(const int* __restrict__ srcIdx, const int* __restrict__ tgtIdx,
                           const float* __restrict__ sc, const float* __restrict__ denom,
                           const float* __restrict__ v, const float* __restrict__ ep,
                           float* __restrict__ agg, long total) {
    long gid = (long)blockIdx.x * blockDim.x + threadIdx.x;
    if (gid >= total) return;
    int e = (int)(gid >> 4);
    int d = (int)(gid & 15) << 2;   // 0,4,...,60
    int h = d >> 4;
    int s = srcIdx[e], t = tgtIdx[e];
    float alpha = sc[(size_t)e * 4 + h] / (denom[(size_t)t * 4 + h] + 1e-9f);
    float4 vv = *(const float4*)(v  + (size_t)s * 64 + d);
    float4 ee = *(const float4*)(ep + (size_t)e * 64 + d);
    float* ap = agg + (size_t)t * 64 + d;
    atomicAdd(ap + 0, alpha * (vv.x + ee.x));
    atomicAdd(ap + 1, alpha * (vv.y + ee.y));
    atomicAdd(ap + 2, alpha * (vv.z + ee.z));
    atomicAdd(ap + 3, alpha * (vv.w + ee.w));
}

// new_nodes = x + agg @ Wo  (WMMA + residual epilogue); wfrag -> Wo fragments
__global__ void outproj_kernel(const float* __restrict__ x, const float* __restrict__ agg,
                               const __bf16* __restrict__ wfrag, float* __restrict__ nn,
                               int ntiles) {
    int lane = threadIdx.x & 31;
    int tile = blockIdx.x * (blockDim.x >> 5) + (threadIdx.x >> 5);
    if (tile >= ntiles) return;
    int m0 = tile << 4;
    const float* rowPtr = agg + (size_t)(m0 + (lane & 15)) * 64;

    v16bf a0 = load_row_bf16(rowPtr, 0, lane);
    v16bf a1 = load_row_bf16(rowPtr, 32, lane);

    v8f acc[4];
#pragma unroll
    for (int i = 0; i < 4; ++i) acc[i] = V8F_ZERO;
#pragma unroll
    for (int t = 0; t < 4; ++t) {
        acc[t] = WMMA_BF16(a0, load_frag(wfrag, t * 2 + 0, lane), acc[t]);
        acc[t] = WMMA_BF16(a1, load_frag(wfrag, t * 2 + 1, lane), acc[t]);
    }
    int rbase = m0 + ((lane >> 4) << 3);
#pragma unroll
    for (int t = 0; t < 4; ++t) {
        int col = (t << 4) + (lane & 15);
#pragma unroll
        for (int j = 0; j < 8; ++j) {
            size_t idx = (size_t)(rbase + j) * 64 + col;
            nn[idx] = acc[t][j] + x[idx];
        }
    }
}

// edge classifier: gather [nn[src] | nn[tgt] | ef] (16x192 bf16 A-tile),
// h = GELU(A @ W1 + b1) (WMMA, 6 K-steps x 4 N-tiles), then 64->2 head via LDS.
// wfrag -> W1 fragments (24: frag = t*6 + s).
__global__ void edge_mlp_kernel(const float* __restrict__ nn, const float* __restrict__ ef,
                                const int* __restrict__ srcIdx, const int* __restrict__ tgtIdx,
                                const __bf16* __restrict__ wfrag, const float* __restrict__ b1,
                                const float* __restrict__ W2, const float* __restrict__ b2,
                                float* __restrict__ out, int etiles) {
    __shared__ float hbuf[8][16][65];  // [wave][edge-in-tile][hidden], padded
    int lane = threadIdx.x & 31;
    int wave = threadIdx.x >> 5;
    int tile = blockIdx.x * (blockDim.x >> 5) + wave;
    bool active = tile < etiles;
    int e0 = tile << 4;
    int el = e0 + (lane & 15);

    if (active) {
        __builtin_prefetch(wfrag, 0, 0);  // global_prefetch_b8: warm packed weights
        int sRow = srcIdx[el];
        int tRow = tgtIdx[el];

        v8f acc[4];
#pragma unroll
        for (int i = 0; i < 4; ++i) acc[i] = V8F_ZERO;

#pragma unroll
        for (int s = 0; s < 6; ++s) {
            int ks = s << 5;
            const float* rowPtr;
            int ofs;
            if (ks < 64)       { rowPtr = nn + (size_t)sRow * 64; ofs = ks;       }
            else if (ks < 128) { rowPtr = nn + (size_t)tRow * 64; ofs = ks - 64;  }
            else               { rowPtr = ef + (size_t)el * 64;   ofs = ks - 128; }
            v16bf a = load_row_bf16(rowPtr, ofs, lane);
#pragma unroll
            for (int t = 0; t < 4; ++t)
                acc[t] = WMMA_BF16(a, load_frag(wfrag, t * 6 + s, lane), acc[t]);
        }
        // bias + exact GELU epilogue into LDS
        int rbase = (lane >> 4) << 3;
#pragma unroll
        for (int t = 0; t < 4; ++t) {
            int col = (t << 4) + (lane & 15);
            float bb = b1[col];
#pragma unroll
            for (int j = 0; j < 8; ++j) {
                float xv = acc[t][j] + bb;
                hbuf[wave][rbase + j][col] =
                    0.5f * xv * (1.0f + erff(xv * 0.70710678118654752f));
            }
        }
    }
    __syncthreads();
    if (active) {
        int er = lane >> 1;   // edge within tile (0..15)
        int c  = lane & 1;    // output class
        float s = b2[c];
#pragma unroll
        for (int d = 0; d < 64; ++d) s += hbuf[wave][er][d] * W2[d * 2 + c];
        out[(size_t)(e0 + er) * 2 + c] = s;
    }
}

// ---------------------------------------------------------------------------
extern "C" void kernel_launch(void* const* d_in, const int* in_sizes, int n_in,
                              void* d_out, int out_size, void* d_ws, size_t ws_size,
                              hipStream_t stream) {
    const float* x    = (const float*)d_in[0];   // node_features [N,64]
    const float* ef   = (const float*)d_in[1];   // edge_features [E,64]
    const int*   eidx = (const int*)d_in[2];     // edge_index [2,E]
    const float* Wq   = (const float*)d_in[3];
    const float* Wk   = (const float*)d_in[4];
    const float* Wv   = (const float*)d_in[5];
    const float* We   = (const float*)d_in[6];
    const float* Wo   = (const float*)d_in[7];
    const float* W1   = (const float*)d_in[8];   // [192,64]
    const float* b1   = (const float*)d_in[9];   // [64]
    const float* W2   = (const float*)d_in[10];  // [64,2]
    const float* b2   = (const float*)d_in[11];  // [2]
    float*       out  = (float*)d_out;           // [E,2]

    const int N = in_sizes[0] / 64;
    const int E = in_sizes[1] / 64;
    const int* srcIdx = eidx;
    const int* tgtIdx = eidx + E;

    // --- workspace layout (f32 words; every region 32B-aligned) ---
    float* ws = (float*)d_ws;
    size_t o = 0;
    float* q   = ws + o; o += (size_t)N * 64;
    float* k   = ws + o; o += (size_t)N * 64;
    float* v   = ws + o; o += (size_t)N * 64;
    float* nn  = ws + o; o += (size_t)N * 64;
    float* ep  = ws + o; o += (size_t)E * 64;
    float* sc  = ws + o; o += (size_t)E * 4;
    unsigned* smaxU = (unsigned*)(ws + o);        // N*4 words
    float*    denom = ws + o + (size_t)N * 4;     // N*4 words
    float*    agg   = ws + o + (size_t)N * 8;     // N*64 words
    long zeroWords  = (long)N * 72;               // smaxU + denom + agg
    o += (size_t)N * 72;
    __bf16* wpack = (__bf16*)(ws + o);            // 64 frags * 512 bf16 = 16384 words

    const int THREADS = 256;  // 8 waves per block on wave32

    // 1) zero smax/denom/agg (must happen every call for graph replay)
    zero_kernel<<<(int)((zeroWords + THREADS - 1) / THREADS), THREADS, 0, stream>>>(
        smaxU, zeroWords);

    // 2) pack weights into fragment-ordered bf16 (64 fragments = 64 waves)
    pack_weights_kernel<<<8, THREADS, 0, stream>>>(Wq, Wk, Wv, We, Wo, W1, wpack);

    // 3) q,k,v projections (fragment base folded into the pointer -> saddr+imm)
    int ntilesN = N / 16;
    gemm64_kernel<<<(ntilesN + 7) / 8, THREADS, 0, stream>>>(x, wpack +  0 * 512, q, ntilesN);
    gemm64_kernel<<<(ntilesN + 7) / 8, THREADS, 0, stream>>>(x, wpack +  8 * 512, k, ntilesN);
    gemm64_kernel<<<(ntilesN + 7) / 8, THREADS, 0, stream>>>(x, wpack + 16 * 512, v, ntilesN);

    // 4) edge projection ep = ef @ We
    int ntilesE = E / 16;
    gemm64_kernel<<<(ntilesE + 7) / 8, THREADS, 0, stream>>>(ef, wpack + 24 * 512, ep, ntilesE);

    // 5) scores + segment max
    int totEH = E * 4;
    scores_kernel<<<(totEH + THREADS - 1) / THREADS, THREADS, 0, stream>>>(
        q, k, ep, srcIdx, tgtIdx, sc, smaxU, totEH);

    // 6) exp + denom
    exp_kernel<<<(totEH + THREADS - 1) / THREADS, THREADS, 0, stream>>>(
        tgtIdx, smaxU, sc, denom, totEH);

    // 7) weighted message scatter into agg (float4 per thread)
    long totED = (long)E * 16;
    msg_kernel<<<(int)((totED + THREADS - 1) / THREADS), THREADS, 0, stream>>>(
        srcIdx, tgtIdx, sc, denom, v, ep, agg, totED);

    // 8) new_nodes = x + agg @ Wo
    outproj_kernel<<<(ntilesN + 7) / 8, THREADS, 0, stream>>>(
        x, agg, wpack + 32 * 512, nn, ntilesN);

    // 9) edge MLP classifier
    edge_mlp_kernel<<<(ntilesE + 7) / 8, THREADS, 0, stream>>>(
        nn, ef, srcIdx, tgtIdx, wpack + 40 * 512, b1, W2, b2, out, ntilesE);
}
